// RKT_41927470744223
// MI455X (gfx1250) — matrix operand order repaired and checked by reading, not compiled, not run on hardware
//
#include <hip/hip_runtime.h>
#include <hip/hip_bf16.h>
#include <stdint.h>

// ---------------- problem constants ----------------
#define Bc   4
#define Sc   2048
#define Ec   128
#define Hc   4
#define HDc  32
#define BSc  (Bc * Sc)          // 8192 rows

typedef __attribute__((ext_vector_type(16))) __bf16 v16bf;
typedef __attribute__((ext_vector_type(8)))  __bf16 v8bf;
typedef __attribute__((ext_vector_type(8)))  float  v8f;

__device__ __forceinline__ __bf16 f2bf(float f) {
  union { float f; uint32_t u; } in; in.f = f;
  uint32_t u = in.u;
  uint32_t r = (u + 0x7fffu + ((u >> 16) & 1u)) >> 16;   // round-nearest-even
  union { unsigned short s; __bf16 b; } out; out.s = (unsigned short)r;
  return out.b;
}

__device__ __forceinline__ float bf2f(__bf16 b) {
  union { unsigned short s; __bf16 b; } in; in.b = b;
  union { uint32_t u; float f; } out; out.u = ((uint32_t)in.s) << 16;
  return out.f;
}

__device__ __forceinline__ v16bf cat8(v8bf lo, v8bf hi) {
  v16bf r;
#pragma unroll
  for (int i = 0; i < 8; ++i) { r[i] = lo[i]; r[8 + i] = hi[i]; }
  return r;
}

// ---------------- weight convert + transpose: Wt[n*K+k] = (bf16)W[k*N+n] ----
__global__ void convT_kernel(const float* __restrict__ W, __bf16* __restrict__ Wt,
                             int K, int N) {
  int idx = blockIdx.x * blockDim.x + threadIdx.x;
  if (idx >= K * N) return;
  int k = idx / N, n = idx % N;
  Wt[(size_t)n * K + k] = f2bf(W[idx]);
}

// ---------------- feature assembly -----------------------------------------
__global__ void assemble_kernel(const int* __restrict__ item_in,
                                const int* __restrict__ skill_in,
                                const int* __restrict__ label_in,
                                const int* __restrict__ item_ids,
                                const int* __restrict__ skill_ids,
                                const float* __restrict__ item_emb,
                                const float* __restrict__ skill_emb,
                                __bf16* __restrict__ Xcat,
                                __bf16* __restrict__ Qcat) {
  int idx = blockIdx.x * blockDim.x + threadIdx.x;   // over BS*128
  if (idx >= BSc * Ec) return;
  int row = idx / Ec, e = idx % Ec;
  float y = (float)label_in[row];
  float iv = item_emb[(size_t)item_in[row] * Ec + e];
  float sv = skill_emb[(size_t)skill_in[row] * Ec + e];
  __bf16* xr = Xcat + (size_t)row * (4 * Ec);
  xr[e]           = f2bf(iv * y);
  xr[Ec + e]      = f2bf(iv * (1.0f - y));
  xr[2 * Ec + e]  = f2bf(sv * y);
  xr[3 * Ec + e]  = f2bf(sv * (1.0f - y));
  __bf16* qr = Qcat + (size_t)row * (2 * Ec);
  qr[e]      = f2bf(item_emb[(size_t)item_ids[row] * Ec + e]);
  qr[Ec + e] = f2bf(skill_emb[(size_t)skill_ids[row] * Ec + e]);
}

// ---------------- generic bf16 WMMA GEMM:  C = act(A[M,K] * Bt[N,K]^T + bias)
__global__ void wmma_gemm_kernel(const __bf16* __restrict__ A,
                                 const __bf16* __restrict__ Bt,
                                 const float* __restrict__ bias,
                                 __bf16* __restrict__ C,
                                 int M, int Kd, int N, int do_relu) {
  int tiles_n = N >> 4;
  int tm = blockIdx.x / tiles_n;
  int tn = blockIdx.x % tiles_n;
  int lane = threadIdx.x & 31;
  int lm = lane & 15;
  int hi = (lane >> 4) & 1;            // 0: lanes 0-15, 1: lanes 16-31
  int abase = hi ? 16 : 0;             // A-matrix contiguous k = 16h + s (matches B)
  int bkoff = hi ? 16 : 0;             // B-matrix 16-bit layout
  int row = tm * 16 + lm;
  int col = tn * 16 + lm;
  const __bf16* arow = A + (size_t)row * Kd;
  const __bf16* brow = Bt + (size_t)col * Kd;

  v8f c = {};
  for (int k = 0; k < Kd; k += 32) {
    __builtin_prefetch(arow + k + 128, 0, 1);
    v8bf alo = *(const v8bf*)(arow + k + abase);
    v8bf ahi = *(const v8bf*)(arow + k + abase + 8);
    v16bf av = cat8(alo, ahi);
    v16bf bv = *(const v16bf*)(brow + k + bkoff);
    c = __builtin_amdgcn_wmma_f32_16x16x32_bf16(false, av, false, bv,
                                                (short)0, c, false, false);
  }
  float bcol = bias[col];
#pragma unroll
  for (int v = 0; v < 8; ++v) {
    int m = tm * 16 + v + (hi ? 8 : 0);
    float val = c[v] + bcol;
    if (do_relu) val = fmaxf(val, 0.0f);
    C[(size_t)m * N + col] = f2bf(val);
  }
}

// ---------------- V transpose to [B,H,HD,S] ---------------------------------
__global__ void transposeV_kernel(const __bf16* __restrict__ Vm,
                                  __bf16* __restrict__ Vt) {
  int idx = blockIdx.x * blockDim.x + threadIdx.x;   // over BS*128
  if (idx >= BSc * Ec) return;
  int row = idx / Ec, dc = idx % Ec;
  int b = row / Sc, s = row % Sc;
  int h = dc / HDc, d = dc % HDc;
  Vt[((size_t)((b * Hc + h) * HDc + d)) * Sc + s] = Vm[idx];
}

// ---------------- fused attention ------------------------------------------
// grid: B*H*(S/16) blocks, 256 threads (8 waves)
// dyn smem: sc f32[16][2048] | pA bf16[16][2048] | ot f32[16][32]
#define SMEM_SC_BYTES  (16 * Sc * 4)
#define SMEM_PA_BYTES  (16 * Sc * 2)
#define SMEM_OT_BYTES  (16 * HDc * 4)
#define SMEM_TOTAL     (SMEM_SC_BYTES + SMEM_PA_BYTES + SMEM_OT_BYTES)

__global__ void attention_kernel(const __bf16* __restrict__ Qm,   // [BS,128]
                                 const __bf16* __restrict__ Km,   // [BS,128]
                                 const __bf16* __restrict__ Vt,   // [B,H,HD,S]
                                 const float* __restrict__ rel,   // [B,S,S]
                                 const float* __restrict__ ts,    // [B,S,S]
                                 const float* __restrict__ l1p,
                                 const float* __restrict__ l2p,
                                 float* __restrict__ probOut,     // [B,H,S,S]
                                 float* __restrict__ outbuf) {    // [BS,128]
  extern __shared__ char smem[];
  float*  sc = (float*)smem;
  __bf16* pA = (__bf16*)(smem + SMEM_SC_BYTES);
  float*  ot = (float*)(smem + SMEM_SC_BYTES + SMEM_PA_BYTES);

  int bh   = blockIdx.x / (Sc / 16);
  int tile = blockIdx.x % (Sc / 16);
  int b = bh / Hc, h = bh % Hc;
  int i0 = tile * 16;

  int w    = threadIdx.x >> 5;        // wave id 0..7
  int lane = threadIdx.x & 31;
  int lm   = lane & 15;
  int hihf = (lane >> 4) & 1;
  int abase = hihf ? 16 : 0;
  int bkoff = hihf ? 16 : 0;
  const float scale = 0.17677669529663687f;   // 1/sqrt(32)

  // -------- phase 1: scores = (Q K^T) * scale via WMMA --------------------
  {
    const __bf16* qp = Qm + (size_t)(b * Sc + i0 + lm) * Ec + h * HDc;
    v16bf aQ = cat8(*(const v8bf*)(qp + abase), *(const v8bf*)(qp + abase + 8));
    for (int kt = w; kt < Sc / 16; kt += 8) {
      int key = kt * 16 + lm;
      const __bf16* kp = Km + (size_t)(b * Sc + key) * Ec + h * HDc + bkoff;
      __builtin_prefetch(kp + 128 * Ec, 0, 1);
      v16bf bK = *(const v16bf*)kp;
      v8f c = {};
      c = __builtin_amdgcn_wmma_f32_16x16x32_bf16(false, aQ, false, bK,
                                                  (short)0, c, false, false);
#pragma unroll
      for (int v = 0; v < 8; ++v) {
        int m = v + (hihf ? 8 : 0);
        sc[m * Sc + kt * 16 + lm] = c[v] * scale;
      }
    }
  }
  __syncthreads();

  // -------- phase 2: triple softmax + blend, stream prob_attn -------------
  // pass1: maxes; stash tv=exp(-|ts|) as bf16 in pA (ts read ONCE)
  // pass2: sums;  sc <- e_p,  pA <- e_t
  // pass3: finals; NT-store prob_attn, pA <- bf16(prob) for the PV WMMA
  float l1v = l1p[0], l2v = l2p[0];
  for (int rr = 0; rr < 2; ++rr) {
    int m  = w * 2 + rr;
    int qi = i0 + m;
    const float* relRow = rel + ((size_t)b * Sc + qi) * Sc;
    const float* tsRow  = ts  + ((size_t)b * Sc + qi) * Sc;

    float pmax = -__builtin_inff(), tmax = -__builtin_inff(), rmax = -__builtin_inff();
    for (int j = lane; j < Sc; j += 32) {
      bool fut = j > qi;
      float s  = fut ? -1e9f : sc[m * Sc + j];
      float tv = fut ? -__builtin_inff() : expf(-fabsf(tsRow[j]));
      pA[m * Sc + j] = f2bf(tv);
      float rvr = relRow[j];
      float rv = (fut && rvr != 0.0f) ? rvr : -10000.0f;
      pmax = fmaxf(pmax, s); tmax = fmaxf(tmax, tv); rmax = fmaxf(rmax, rv);
    }
#pragma unroll
    for (int o = 16; o > 0; o >>= 1) {
      pmax = fmaxf(pmax, __shfl_xor(pmax, o, 32));
      tmax = fmaxf(tmax, __shfl_xor(tmax, o, 32));
      rmax = fmaxf(rmax, __shfl_xor(rmax, o, 32));
    }
    float psum = 0.f, tsum = 0.f, rsum = 0.f;
    for (int j = lane; j < Sc; j += 32) {
      int idx = m * Sc + j;
      bool fut = j > qi;
      float s  = fut ? -1e9f : sc[idx];
      float ep = expf(s - pmax);
      sc[idx] = ep; psum += ep;
      float et = expf(bf2f(pA[idx]) - tmax);
      pA[idx] = f2bf(et); tsum += et;
      float rvr = relRow[j];
      float rv = (fut && rvr != 0.0f) ? rvr : -10000.0f;
      rsum += expf(rv - rmax);
    }
#pragma unroll
    for (int o = 16; o > 0; o >>= 1) {
      psum += __shfl_xor(psum, o, 32);
      tsum += __shfl_xor(tsum, o, 32);
      rsum += __shfl_xor(rsum, o, 32);
    }
    float* pout = probOut + ((size_t)(b * Hc + h) * Sc + qi) * Sc;
    float pinv = 1.0f / psum, tinv = 1.0f / tsum, rinv = 1.0f / rsum;
    for (int j = lane; j < Sc; j += 32) {
      int idx = m * Sc + j;
      bool fut = j > qi;
      float p = sc[idx] * pinv;
      float t = bf2f(pA[idx]) * tinv;
      float rvr = relRow[j];
      float rv = (fut && rvr != 0.0f) ? rvr : -10000.0f;
      float r = expf(rv - rmax) * rinv;
      float f = (1.0f - l1v) * ((1.0f - l2v) * p + l2v * t) + l1v * r;
      __builtin_nontemporal_store(f, pout + j);   // NT: keep rel/ts in L2
      pA[idx] = f2bf(f);
    }
  }
  __syncthreads();

  for (int idx = threadIdx.x; idx < 16 * HDc; idx += 256) ot[idx] = 0.0f;
  __syncthreads();

  // -------- phase 3: out = prob @ V via WMMA ------------------------------
  {
    v8f c0 = {}, c1 = {};
    for (int kc = w; kc < Sc / 32; kc += 8) {
      const __bf16* ap = pA + lm * Sc + kc * 32 + abase;
      v16bf aP = cat8(*(const v8bf*)ap, *(const v8bf*)(ap + 8));
      const __bf16* vp0 =
          Vt + ((size_t)((b * Hc + h) * HDc) + lm) * Sc + kc * 32 + bkoff;
      const __bf16* vp1 = vp0 + (size_t)16 * Sc;
      __builtin_prefetch(vp0 + 256, 0, 1);
      v16bf bV0 = *(const v16bf*)vp0;
      v16bf bV1 = *(const v16bf*)vp1;
      c0 = __builtin_amdgcn_wmma_f32_16x16x32_bf16(false, aP, false, bV0,
                                                   (short)0, c0, false, false);
      c1 = __builtin_amdgcn_wmma_f32_16x16x32_bf16(false, aP, false, bV1,
                                                   (short)0, c1, false, false);
    }
#pragma unroll
    for (int v = 0; v < 8; ++v) {
      int m = v + (hihf ? 8 : 0);
      atomicAdd(&ot[m * HDc + lm],      c0[v]);
      atomicAdd(&ot[m * HDc + 16 + lm], c1[v]);
    }
  }
  __syncthreads();

  for (int idx = threadIdx.x; idx < 16 * HDc; idx += 256) {
    int m = idx / HDc, d = idx % HDc;
    outbuf[((size_t)(b * Sc + i0 + m)) * Ec + h * HDc + d] = ot[idx];
  }
}

// ---------------- final projection pred = out @ Wout + bout -----------------
__global__ void pred_kernel(const float* __restrict__ outbuf,
                            const float* __restrict__ Wout,
                            const float* __restrict__ bout,
                            float* __restrict__ pred) {
  int row = blockIdx.x * blockDim.x + threadIdx.x;
  if (row >= BSc) return;
  float acc = bout[0];
  const float* r = outbuf + (size_t)row * Ec;
#pragma unroll 4
  for (int d = 0; d < Ec; ++d) acc += r[d] * Wout[d];
  pred[row] = acc;
}

// ---------------- host side --------------------------------------------------
extern "C" void kernel_launch(void* const* d_in, const int* in_sizes, int n_in,
                              void* d_out, int out_size, void* d_ws, size_t ws_size,
                              hipStream_t stream) {
  (void)in_sizes; (void)n_in; (void)out_size; (void)ws_size;
  const int*   item_in  = (const int*)d_in[0];
  const int*   skill_in = (const int*)d_in[1];
  const int*   label_in = (const int*)d_in[2];
  const int*   item_ids = (const int*)d_in[3];
  const int*   skill_ids= (const int*)d_in[4];
  const float* rel      = (const float*)d_in[5];
  const float* tsm      = (const float*)d_in[6];
  const float* item_emb = (const float*)d_in[7];
  const float* skill_emb= (const float*)d_in[8];
  const float* Win      = (const float*)d_in[9];
  const float* b_in     = (const float*)d_in[10];
  const float* Wq       = (const float*)d_in[11];
  const float* bq       = (const float*)d_in[12];
  const float* Wk       = (const float*)d_in[13];
  const float* bk       = (const float*)d_in[14];
  const float* Wv       = (const float*)d_in[15];
  const float* bv       = (const float*)d_in[16];
  const float* Wout     = (const float*)d_in[17];
  const float* bout     = (const float*)d_in[18];
  const float* l1       = (const float*)d_in[19];
  const float* l2       = (const float*)d_in[20];

  float* pred    = (float*)d_out;            // [8192]
  float* probOut = (float*)d_out + BSc;      // [B,H,S,S]

  // workspace carve-up (256B aligned)
  char* ws = (char*)d_ws;
  size_t off = 0;
  auto alloc = [&](size_t bytes) { char* p = ws + off; off = (off + bytes + 255) & ~(size_t)255; return p; };
  __bf16* WinT  = (__bf16*)alloc((size_t)4 * Ec * 2 * Ec * 2);   // [256,512]
  __bf16* WqT   = (__bf16*)alloc((size_t)2 * Ec * Ec * 2);       // [128,256]
  __bf16* WkT   = (__bf16*)alloc((size_t)2 * Ec * Ec * 2);
  __bf16* WvT   = (__bf16*)alloc((size_t)2 * Ec * Ec * 2);
  __bf16* Xcat  = (__bf16*)alloc((size_t)BSc * 4 * Ec * 2);      // [8192,512]
  __bf16* Qcat  = (__bf16*)alloc((size_t)BSc * 2 * Ec * 2);      // [8192,256]
  __bf16* inpB  = (__bf16*)alloc((size_t)BSc * 2 * Ec * 2);      // [8192,256]
  __bf16* Qm    = (__bf16*)alloc((size_t)BSc * Ec * 2);          // [8192,128]
  __bf16* Km    = (__bf16*)alloc((size_t)BSc * Ec * 2);
  __bf16* Vm    = (__bf16*)alloc((size_t)BSc * Ec * 2);
  __bf16* Vt    = (__bf16*)alloc((size_t)BSc * Ec * 2);          // [B,H,HD,S]
  float*  outbuf= (float*)alloc((size_t)BSc * Ec * 4);           // [8192,128]

  convT_kernel<<<(4 * Ec * 2 * Ec + 255) / 256, 256, 0, stream>>>(Win, WinT, 4 * Ec, 2 * Ec);
  convT_kernel<<<(2 * Ec * Ec + 255) / 256, 256, 0, stream>>>(Wq, WqT, 2 * Ec, Ec);
  convT_kernel<<<(2 * Ec * Ec + 255) / 256, 256, 0, stream>>>(Wk, WkT, 2 * Ec, Ec);
  convT_kernel<<<(2 * Ec * Ec + 255) / 256, 256, 0, stream>>>(Wv, WvT, 2 * Ec, Ec);

  assemble_kernel<<<(BSc * Ec + 255) / 256, 256, 0, stream>>>(
      item_in, skill_in, label_in, item_ids, skill_ids, item_emb, skill_emb, Xcat, Qcat);

  wmma_gemm_kernel<<<(BSc / 16) * (2 * Ec / 16), 32, 0, stream>>>(
      Xcat, WinT, b_in, inpB, BSc, 4 * Ec, 2 * Ec, 1);
  wmma_gemm_kernel<<<(BSc / 16) * (Ec / 16), 32, 0, stream>>>(
      Qcat, WqT, bq, Qm, BSc, 2 * Ec, Ec, 0);
  wmma_gemm_kernel<<<(BSc / 16) * (Ec / 16), 32, 0, stream>>>(
      inpB, WkT, bk, Km, BSc, 2 * Ec, Ec, 0);
  wmma_gemm_kernel<<<(BSc / 16) * (Ec / 16), 32, 0, stream>>>(
      inpB, WvT, bv, Vm, BSc, 2 * Ec, Ec, 0);

  transposeV_kernel<<<(BSc * Ec + 255) / 256, 256, 0, stream>>>(Vm, Vt);

  attention_kernel<<<Bc * Hc * (Sc / 16), 256, SMEM_TOTAL, stream>>>(
      Qm, Km, Vt, rel, tsm, l1, l2, probOut, outbuf);

  pred_kernel<<<(BSc + 255) / 256, 256, 0, stream>>>(outbuf, Wout, bout, pred);
}
